// Attention_73907797229897
// MI455X (gfx1250) — matrix operand order repaired and checked by reading, not compiled, not run on hardware
//
#include <hip/hip_runtime.h>

// ---------------------------------------------------------------------------
// MI455X (gfx1250, wave32) fused attention-scores kernel.
//   Kernel 1: q = bf16(x @ Wq + bq)         -> d_ws   (v_wmma_f32_16x16x32_bf16)
//   Kernel 2: s = softmax(q q^T / 8) per (b,h): scores staged in LDS, k tiles
//             streamed with double-buffered GLOBAL_LOAD_ASYNC_TO_LDS_B128.
// ---------------------------------------------------------------------------

#define HIDDEN 1024
#define SEQ    2048
#define NHEADS 16
#define HDIM   64

typedef __attribute__((ext_vector_type(16))) __bf16 bf16x16;
typedef __attribute__((ext_vector_type(8)))  float  f32x8;

// Two 16B LDS chunks -> one 32B bf16 WMMA fragment (ds_load_b128 x2).
__device__ __forceinline__ bf16x16 ld_frag(const __bf16* rowPtr, int off0, int off1) {
    union { uint4 u[2]; bf16x16 v; } f;
    f.u[0] = *(const uint4*)(rowPtr + off0);
    f.u[1] = *(const uint4*)(rowPtr + off1);
    return f.v;
}

// Async DMA one 16B chunk: global memory -> LDS (ASYNCcnt-tracked, no VGPR data).
// Generic LDS pointers carry the wave-relative AS(3) offset in the low 32 bits,
// which is exactly the VDST operand of global_load_async_to_lds_*.
__device__ __forceinline__ void async_copy16(const __bf16* gsrc, __bf16* ldst) {
    unsigned           lds_off = (unsigned)(unsigned long long)(const void*)ldst;
    unsigned long long gaddr   = (unsigned long long)(const void*)gsrc;
    asm volatile("global_load_async_to_lds_b128 %0, %1, off"
                 :: "v"(lds_off), "v"(gaddr) : "memory");
}
__device__ __forceinline__ void wait_async_le(void) {   // next tile (4) may remain
    asm volatile("s_wait_asynccnt 4" ::: "memory");
}
__device__ __forceinline__ void wait_async_0(void) {
    asm volatile("s_wait_asynccnt 0" ::: "memory");
}

// ---------------------------------------------------------------------------
// Kernel 1: q projection. Grid (N/128, M/64), 256 threads (8 waves).
// 64(M) x 128(N) f32 accum tile, K step 32. x -> bf16 LDS; Wq transposed [n][k].
// ---------------------------------------------------------------------------
__global__ __launch_bounds__(256)
void qproj_kernel(const float* __restrict__ x, const float* __restrict__ Wq,
                  const float* __restrict__ bq, __bf16* __restrict__ qb)
{
    __shared__ __bf16 sA[64][32];    // x tile   [m][k]        (64B rows)
    __shared__ __bf16 sB[128][40];   // Wq^T tile[n][k] padded (80B rows)

    const int tid  = threadIdx.x;
    const int lane = tid & 31;
    const int wave = tid >> 5;
    const int m0   = blockIdx.y * 64;
    const int n0   = blockIdx.x * 128;

    const int mt     = wave >> 1;        // 4 row tiles, 2 waves per row tile
    const int ntBase = (wave & 1) * 4;   // 4 col tiles per wave

    f32x8 acc[4] = {};

    const int hi8  = (lane >> 4) * 8;    // A-fragment K split (ISA 16-bit A layout)
    const int hi16 = (lane >> 4) * 16;   // unused: B fragments also use hi8

    for (int k0 = 0; k0 < HIDDEN; k0 += 32) {
        // --- stage A: 64x32 f32 -> bf16, coalesced 32B/lane
        {
            const int row = tid >> 2;
            const int kp  = (tid & 3) * 8;
            const float* src = x + (size_t)(m0 + row) * HIDDEN + k0 + kp;
            float4 v0 = *(const float4*)(src);
            float4 v1 = *(const float4*)(src + 4);
            if (k0 + 32 < HIDDEN)                     // global_prefetch_b8
                __builtin_prefetch(src + 32, 0, 0);
            __bf16* dst = &sA[row][kp];
            dst[0] = (__bf16)v0.x; dst[1] = (__bf16)v0.y;
            dst[2] = (__bf16)v0.z; dst[3] = (__bf16)v0.w;
            dst[4] = (__bf16)v1.x; dst[5] = (__bf16)v1.y;
            dst[6] = (__bf16)v1.z; dst[7] = (__bf16)v1.w;
        }
        // --- stage B: Wq[k][n] read coalesced, written transposed sB[n][k]
        {
            const int kk = tid >> 3;
            const int np = (tid & 7) * 16;
            const float* src = Wq + (size_t)(k0 + kk) * HIDDEN + n0 + np;
            #pragma unroll
            for (int j = 0; j < 16; j += 4) {
                float4 v = *(const float4*)(src + j);
                sB[np + j + 0][kk] = (__bf16)v.x;
                sB[np + j + 1][kk] = (__bf16)v.y;
                sB[np + j + 2][kk] = (__bf16)v.z;
                sB[np + j + 3][kk] = (__bf16)v.w;
            }
        }
        __syncthreads();

        // Preload all fragments, then issue WMMAs back-to-back.
        bf16x16 aF = ld_frag(&sA[mt * 16 + (lane & 15)][0], hi8, 16 + hi8);
        bf16x16 bF[4];
        #pragma unroll
        for (int i = 0; i < 4; ++i)
            bF[i] = ld_frag(&sB[(ntBase + i) * 16 + (lane & 15)][0], hi8, 16 + hi8);
        #pragma unroll
        for (int i = 0; i < 4; ++i)
            acc[i] = __builtin_amdgcn_wmma_f32_16x16x32_bf16(
                         false, aF, false, bF[i], (short)0, acc[i], false, false);
        __syncthreads();
    }

    // epilogue: + bias, convert to bf16, scatter per ISA C/D layout
    #pragma unroll
    for (int i = 0; i < 4; ++i) {
        const int nt = ntBase + i;
        const int n  = n0 + nt * 16 + (lane & 15);
        const float bias = bq[n];
        #pragma unroll
        for (int r = 0; r < 8; ++r) {
            const int m = m0 + mt * 16 + r + 8 * (lane >> 4);
            qb[(size_t)m * HIDDEN + n] = (__bf16)(acc[i][r] + bias);
        }
    }
}

// ---------------------------------------------------------------------------
// Kernel 2: fused scores + softmax. Grid (S/32, H, B), 256 threads (8 waves).
// LDS: 32 x 2052 f32 score strip + 32x80 bf16 q strip + 2 x (128x80) k tiles.
// ---------------------------------------------------------------------------
#define SSTRIDE 2052
#define SQ_OFF  (32 * SSTRIDE * 4)              // 262656 B
#define SK_OFF  (SQ_OFF + 32 * 80 * 2)          // 267776 B
#define KBUF_ELEMS (128 * 80)
#define SMEM_BYTES (SK_OFF + 2 * KBUF_ELEMS * 2)   // 308736 B (< 320 KB / WGP)
#define NKT (SEQ / 128)

__global__ __launch_bounds__(256)
void attn_kernel(const __bf16* __restrict__ qb, float* __restrict__ out)
{
    extern __shared__ char smem[];
    float*  sS = (float*)smem;
    __bf16* sQ = (__bf16*)(smem + SQ_OFF);
    __bf16* sK = (__bf16*)(smem + SK_OFF);   // two ping-pong k-tile buffers

    const int tid   = threadIdx.x;
    const int lane  = tid & 31;
    const int wave  = tid >> 5;
    const int strip = blockIdx.x;   // 32-row q strip
    const int h     = blockIdx.y;
    const int b     = blockIdx.z;

    const __bf16* qh  = qb + (size_t)b * SEQ * HIDDEN + h * HDIM;
    const int     qr0 = strip * 32;

    // This thread's 4 chunk coordinates within a 128-row k tile (16B chunks).
    int crow[4], cch[4];
    #pragma unroll
    for (int j = 0; j < 4; ++j) {
        const int idx = tid + j * 256;
        crow[j] = idx >> 3;
        cch[j]  = (idx & 7) * 8;
    }

    // Prologue: async-stream k tile 0 into buffer 0.
    #pragma unroll
    for (int j = 0; j < 4; ++j)
        async_copy16(qh + (size_t)crow[j] * HIDDEN + cch[j],
                     sK + crow[j] * 80 + cch[j]);

    // Load q strip: 32 rows x 64 bf16 (one 16B chunk per thread).
    {
        const int row = tid >> 3;
        const int ch  = (tid & 7) * 8;
        *(uint4*)(sQ + row * 80 + ch) =
            *(const uint4*)(qh + (size_t)(qr0 + row) * HIDDEN + ch);
    }
    __syncthreads();

    const int mt   = wave >> 2;        // 2 row tiles
    const int ntB  = (wave & 3) * 2;   // 2 of 8 col tiles per wave
    const int hi8  = (lane >> 4) * 8;
    const int hi16 = (lane >> 4) * 16;
    const int mrow = mt * 16 + (lane & 15);

    // A fragments (q rows) invariant across k tiles: hoist.
    const bf16x16 aF0 = ld_frag(sQ + mrow * 80, 0 + hi8, 16 + hi8);
    const bf16x16 aF1 = ld_frag(sQ + mrow * 80, 32 + hi8, 48 + hi8);

    const float scale = 0.125f;   // 1/sqrt(64)

    for (int kt = 0; kt < NKT; ++kt) {
        __bf16* cur = sK + (kt & 1) * KBUF_ELEMS;
        if (kt + 1 < NKT) {
            // Issue next tile into the other buffer (its last readers synced
            // at the end of iteration kt-1), then wait for THIS tile only.
            __bf16* nxt = sK + ((kt + 1) & 1) * KBUF_ELEMS;
            const __bf16* gsrc = qh + (size_t)(kt + 1) * 128 * HIDDEN;
            #pragma unroll
            for (int j = 0; j < 4; ++j)
                async_copy16(gsrc + (size_t)crow[j] * HIDDEN + cch[j],
                             nxt + crow[j] * 80 + cch[j]);
            wait_async_le();    // s_wait_asynccnt 4 : tile kt landed
        } else {
            wait_async_0();
        }
        __syncthreads();

        // Preload 4 B fragments (2 col tiles x 2 K-halves), then 4 WMMAs.
        const __bf16* kRow0 = cur + (ntB * 16 + (lane & 15)) * 80;
        const __bf16* kRow1 = kRow0 + 16 * 80;
        bf16x16 bF0 = ld_frag(kRow0, 0 + hi8, 16 + hi8);
        bf16x16 bF1 = ld_frag(kRow0, 32 + hi8, 48 + hi8);
        bf16x16 bF2 = ld_frag(kRow1, 0 + hi8, 16 + hi8);
        bf16x16 bF3 = ld_frag(kRow1, 32 + hi8, 48 + hi8);
        f32x8 acc0 = {}, acc1 = {};
        acc0 = __builtin_amdgcn_wmma_f32_16x16x32_bf16(
                   false, aF0, false, bF0, (short)0, acc0, false, false);
        acc0 = __builtin_amdgcn_wmma_f32_16x16x32_bf16(
                   false, aF1, false, bF1, (short)0, acc0, false, false);
        acc1 = __builtin_amdgcn_wmma_f32_16x16x32_bf16(
                   false, aF0, false, bF2, (short)0, acc1, false, false);
        acc1 = __builtin_amdgcn_wmma_f32_16x16x32_bf16(
                   false, aF1, false, bF3, (short)0, acc1, false, false);

        float* sOut = sS + (size_t)(mt * 16 + 8 * (lane >> 4)) * SSTRIDE
                         + kt * 128 + ntB * 16 + (lane & 15);
        #pragma unroll
        for (int r = 0; r < 8; ++r) {
            sOut[r * SSTRIDE]      = acc0[r] * scale;
            sOut[r * SSTRIDE + 16] = acc1[r] * scale;
        }
        __syncthreads();
    }

    // --- softmax: each wave owns 4 rows; lane covers strided float4 columns
    for (int rr = 0; rr < 4; ++rr) {
        const int row = wave * 4 + rr;
        float* rowS = sS + (size_t)row * SSTRIDE;

        float mx = -3.402823466e+38f;
        #pragma unroll
        for (int it = 0; it < 16; ++it) {
            float4 v = *(const float4*)(rowS + it * 128 + lane * 4);
            mx = fmaxf(mx, fmaxf(fmaxf(v.x, v.y), fmaxf(v.z, v.w)));
        }
        #pragma unroll
        for (int off = 16; off > 0; off >>= 1)
            mx = fmaxf(mx, __shfl_xor(mx, off, 32));

        float sum = 0.0f;
        #pragma unroll
        for (int it = 0; it < 16; ++it) {
            float4 v = *(float4*)(rowS + it * 128 + lane * 4);
            v.x = __expf(v.x - mx); v.y = __expf(v.y - mx);
            v.z = __expf(v.z - mx); v.w = __expf(v.w - mx);
            sum += v.x + v.y + v.z + v.w;
            *(float4*)(rowS + it * 128 + lane * 4) = v;
        }
        #pragma unroll
        for (int off = 16; off > 0; off >>= 1)
            sum += __shfl_xor(sum, off, 32);
        const float inv = 1.0f / sum;

        float* dst = out + ((size_t)(b * NHEADS + h) * SEQ + qr0 + row) * SEQ;
        #pragma unroll
        for (int it = 0; it < 16; ++it) {
            float4 v = *(const float4*)(rowS + it * 128 + lane * 4);
            v.x *= inv; v.y *= inv; v.z *= inv; v.w *= inv;
            *(float4*)(dst + it * 128 + lane * 4) = v;   // 512B coalesced/instr
        }
    }
}

// ---------------------------------------------------------------------------
extern "C" void kernel_launch(void* const* d_in, const int* in_sizes, int n_in,
                              void* d_out, int out_size, void* d_ws, size_t ws_size,
                              hipStream_t stream)
{
    const float* x   = (const float*)d_in[0];   // [2,2048,1024] f32
    const float* Wq  = (const float*)d_in[1];   // [1024,1024]   f32
    const float* bq  = (const float*)d_in[2];   // [1024]        f32
    float*       out = (float*)d_out;           // [2,16,2048,2048] f32
    __bf16*      qbf = (__bf16*)d_ws;           // 8 MB scratch: q in bf16

    // Kernel 1: q = bf16(x @ Wq + bq).  M=4096, N=1024, tiles 64x128.
    qproj_kernel<<<dim3(HIDDEN / 128, (2 * SEQ) / 64), 256, 0, stream>>>(x, Wq, bq, qbf);

    // Kernel 2: fused scores + softmax. One block per (b, h, 32-row strip).
    attn_kernel<<<dim3(SEQ / 32, NHEADS, 2), 256, SMEM_BYTES, stream>>>(qbf, out);
}
